// FlashInferBlock_15461882265946
// MI455X (gfx1250) — compile-verified, matrix operand order checked by reading, not run on hardware
//
#include <hip/hip_runtime.h>

// ---------------------------------------------------------------------------
// Types
// ---------------------------------------------------------------------------
typedef __attribute__((ext_vector_type(16))) __bf16        v16bf;
typedef __attribute__((ext_vector_type(8)))  float         v8f;
typedef __attribute__((ext_vector_type(4)))  unsigned int  u32x4;
typedef unsigned short u16;

union Frag {
    u32x4  q[2];
    u16    h[16];
    v16bf  v;
};

__device__ inline v8f vzero8() {
    v8f z;
#pragma unroll
    for (int i = 0; i < 8; ++i) z[i] = 0.0f;
    return z;
}

__device__ inline u16 f2bf(float x) {
    unsigned u = __float_as_uint(x);
    unsigned r = u + 0x7FFFu + ((u >> 16) & 1u);  // round-to-nearest-even
    return (u16)(r >> 16);
}

__device__ inline v8f wmma_bf16(v16bf a, v16bf b, v8f c) {
    return __builtin_amdgcn_wmma_f32_16x16x32_bf16(
        false, a, false, b, (short)0, c, false, false);
}

// Low 32 bits of a generic LDS pointer = LDS byte offset (aperture layout).
__device__ inline unsigned lds_off32(const void* p) {
    return (unsigned)(unsigned long long)p;
}

// CDNA5 async global->LDS DMA (16B per lane), tracked by ASYNCcnt.
__device__ inline void async_copy_b128(unsigned lds_off, const void* gptr) {
    asm volatile("global_load_async_to_lds_b128 %0, %1, off"
                 :: "v"(lds_off), "v"(gptr) : "memory");
}
__device__ inline void wait_async0() {
    asm volatile("s_wait_asynccnt 0x0" ::: "memory");
}

// A-fragment (and N-major B-fragment) loader: element (m, k) at base + m*ld + k.
// Lane l: m = l%16, halves = K {sel*8..sel*8+7, 16+sel*8..+7}, sel = l/16.
__device__ inline v16bf load_frag_row(const u16* base, int ld) {
    int lane = threadIdx.x & 31;
    int sel  = lane >> 4;
    int m    = lane & 15;
    const u16* p = base + m * ld + sel * 8;
    Frag f;
    f.q[0] = *(const u32x4*)(p);        // K = sel*8 .. sel*8+7
    f.q[1] = *(const u32x4*)(p + 16);   // K = 16+sel*8 .. +7
    return f.v;
}

// B-fragment from K-major storage: element (k, n) at base + k*kStride + n.
__device__ inline v16bf load_frag_kmajor(const u16* base, int kStride) {
    int lane = threadIdx.x & 31;
    int sel  = lane >> 4;
    int n    = lane & 15;
    Frag f;
#pragma unroll
    for (int j = 0; j < 8; ++j) {
        f.h[j]     = base[(sel * 8 + j) * kStride + n];
        f.h[j + 8] = base[(16 + sel * 8 + j) * kStride + n];
    }
    return f.v;
}

__device__ inline float redmax16(float v) {
#pragma unroll
    for (int m = 8; m >= 1; m >>= 1) v = fmaxf(v, __shfl_xor(v, m, 16));
    return v;
}
__device__ inline float redsum16(float v) {
#pragma unroll
    for (int m = 8; m >= 1; m >>= 1) v += __shfl_xor(v, m, 16);
    return v;
}

// ---------------------------------------------------------------------------
// LayerNorm: fp32 in -> bf16 out.  One block per row (D elements).
// ---------------------------------------------------------------------------
__global__ void ln_bf16_kernel(const float* __restrict__ x,
                               const float* __restrict__ g,
                               const float* __restrict__ b,
                               u16* __restrict__ out, int D) {
    int row = blockIdx.x;
    const float* xr = x + (size_t)row * D;
    float s = 0.f, s2 = 0.f;
    for (int i = threadIdx.x; i < D; i += blockDim.x) {
        float v = xr[i];
        s += v; s2 += v * v;
    }
#pragma unroll
    for (int m = 16; m >= 1; m >>= 1) {
        s  += __shfl_xor(s,  m, 32);
        s2 += __shfl_xor(s2, m, 32);
    }
    __shared__ float red0[8], red1[8];
    int wv = threadIdx.x >> 5, ln = threadIdx.x & 31;
    if (ln == 0) { red0[wv] = s; red1[wv] = s2; }
    __syncthreads();
    if (threadIdx.x == 0) {
        float a = 0.f, c = 0.f;
        int nw = blockDim.x >> 5;
        for (int i = 0; i < nw; ++i) { a += red0[i]; c += red1[i]; }
        red0[0] = a; red1[0] = c;
    }
    __syncthreads();
    float mean = red0[0] / (float)D;
    float var  = red1[0] / (float)D - mean * mean;
    float rstd = rsqrtf(var + 1e-5f);
    u16* orow = out + (size_t)row * D;
    for (int i = threadIdx.x; i < D; i += blockDim.x) {
        float v = (xr[i] - mean) * rstd * g[i] + b[i];
        orow[i] = f2bf(v);
    }
}

// ---------------------------------------------------------------------------
// Generic bf16 WMMA GEMM:  C[M,N] = A[M,K](bf16) * Bw[K,N](f32->bf16) + bias
// 128x128 block tile, 256 threads (8 waves), wave = 32x64, K-step = 64.
// A tile staged via async global->LDS DMA; B tile converted f32->bf16 in VALU.
// ---------------------------------------------------------------------------
enum { EPI_QKV = 0, EPI_RES = 1, EPI_GELU = 2 };

template <int EPI>
__global__ void gemm_bf16_kernel(const u16*  __restrict__ A,
                                 const float* __restrict__ Bw,
                                 const float* __restrict__ bias,
                                 const float* __restrict__ resid,
                                 u16*   __restrict__ outb,
                                 float* __restrict__ outf,
                                 u16* __restrict__ qout,
                                 u16* __restrict__ kout,
                                 u16* __restrict__ vout,
                                 int M, int N, int K, int S) {
    constexpr int KT  = 64;                 // K-step
    constexpr int LDT = 72;                 // padded LDS stride (halves)
    __shared__ __align__(16) u16 As[128 * LDT];
    __shared__ __align__(16) u16 Bs[128 * LDT];   // N-major (transposed) B tile

    const int t    = threadIdx.x;
    const int bm   = blockIdx.y * 128;
    const int bn   = blockIdx.x * 128;
    const int wv   = t >> 5;
    const int lane = t & 31;
    const int wm   = wv >> 1;               // 0..3  -> 32-row slab
    const int wn   = wv & 1;                // 0..1  -> 64-col slab

    v8f acc[2][4];
#pragma unroll
    for (int mi = 0; mi < 2; ++mi)
#pragma unroll
        for (int ni = 0; ni < 4; ++ni) acc[mi][ni] = vzero8();

    for (int k0 = 0; k0 < K; k0 += KT) {
        // Stage A tile via async DMA: 128 rows x 64 halves = 1024 x 16B chunks.
        for (int c = t; c < 1024; c += 256) {
            int row = c >> 3, cc = c & 7;
            async_copy_b128(lds_off32(As + row * LDT + cc * 8),
                            A + (size_t)(bm + row) * K + k0 + cc * 8);
        }
        // Stage B tile transposed with f32 -> bf16 conversion (coalesced on n).
#pragma unroll
        for (int e = 0; e < 32; ++e) {
            int idx = e * 256 + t;
            int n = idx & 127, kk = idx >> 7;
            float v = Bw[(size_t)(k0 + kk) * N + bn + n];
            Bs[n * LDT + kk] = f2bf(v);
        }
        // Prefetch next weight tile into L2 while we compute.
        if (k0 + KT < K)
            __builtin_prefetch(&Bw[(size_t)(k0 + KT + (t >> 2)) * N + bn + (t & 3) * 32], 0, 1);
        wait_async0();
        __syncthreads();

#pragma unroll
        for (int kk = 0; kk < KT; kk += 32) {
            v16bf af0 = load_frag_row(As + (wm * 32 + 0)  * LDT + kk, LDT);
            v16bf af1 = load_frag_row(As + (wm * 32 + 16) * LDT + kk, LDT);
#pragma unroll
            for (int ni = 0; ni < 4; ++ni) {
                v16bf bfr = load_frag_row(Bs + (wn * 64 + ni * 16) * LDT + kk, LDT);
                acc[0][ni] = wmma_bf16(af0, bfr, acc[0][ni]);
                acc[1][ni] = wmma_bf16(af1, bfr, acc[1][ni]);
            }
        }
        __syncthreads();
    }

    // Epilogue.  C element (m, n): vgpr r -> m = 8*(lane/16)+r, n = lane%16.
    const int mbase = (lane >> 4) * 8;
    const int nloc  = lane & 15;
#pragma unroll
    for (int mi = 0; mi < 2; ++mi) {
#pragma unroll
        for (int ni = 0; ni < 4; ++ni) {
            int gn = bn + wn * 64 + ni * 16 + nloc;
            float bv = bias[gn];
#pragma unroll
            for (int r = 0; r < 8; ++r) {
                int gm = bm + wm * 32 + mi * 16 + mbase + r;
                float v = acc[mi][ni][r] + bv;
                if (EPI == EPI_QKV) {
                    int part = gn >> 10;           // 0=q 1=k 2=v
                    int rem  = gn & 1023;
                    int hh   = rem >> 6;
                    int dd   = rem & 63;
                    int bb   = gm / S;
                    int ss   = gm - bb * S;
                    u16* dst = (part == 0) ? qout : (part == 1) ? kout : vout;
                    dst[(((size_t)bb * 16 + hh) * S + ss) * 64 + dd] = f2bf(v);
                } else if (EPI == EPI_RES) {
                    size_t o = (size_t)gm * N + gn;
                    outf[o] = v + resid[o];
                } else { // EPI_GELU (exact erf form)
                    float gelu = 0.5f * v * (1.0f + erff(v * 0.70710678118654752f));
                    outb[(size_t)gm * N + gn] = f2bf(gelu);
                }
            }
        }
    }
}

// ---------------------------------------------------------------------------
// Flash attention (causal).  Block = (q-tile of 64, head, batch), 128 threads.
// Q,K,V bf16 in [B,H,S,64]; output bf16 [B,S,D] with col = h*64+d.
// Q/K/V tiles staged via async global->LDS DMA.
// ---------------------------------------------------------------------------
__global__ void attn_kernel(const u16* __restrict__ Qb,
                            const u16* __restrict__ Kb,
                            const u16* __restrict__ Vb,
                            u16* __restrict__ Ob, int S) {
    constexpr int LDT = 72;  // padded stride for 64-wide tiles
    __shared__ __align__(16) u16 Qs[64 * LDT];
    __shared__ __align__(16) u16 Ks[64 * LDT];
    __shared__ __align__(16) u16 Vs[64 * LDT];
    __shared__ __align__(16) u16 Ps[64 * 64];

    const int qt = blockIdx.x, h = blockIdx.y, b = blockIdx.z;
    const int t = threadIdx.x, wv = t >> 5, lane = t & 31;
    const int qrow0 = qt * 64;

    const size_t headoff = (((size_t)b * 16 + h) * S) * 64;
    const u16* qh = Qb + headoff;
    const u16* kh = Kb + headoff;
    const u16* vh = Vb + headoff;

    // Stage Q tile (64x64 halves = 512 x 16B; 128 threads x 4) via async DMA.
    for (int c = t; c < 512; c += 128) {
        int row = c >> 3, cc = c & 7;
        async_copy_b128(lds_off32(Qs + row * LDT + cc * 8),
                        qh + (size_t)(qrow0 + row) * 64 + cc * 8);
    }

    float mrow[8], lrow[8];
    v8f o[4];
#pragma unroll
    for (int r = 0; r < 8; ++r) { mrow[r] = -3.0e38f; lrow[r] = 0.0f; }
#pragma unroll
    for (int dt = 0; dt < 4; ++dt) o[dt] = vzero8();

    const float scale = 0.125f;  // 1/sqrt(64)
    const int mbase = (lane >> 4) * 8;
    const int nloc  = lane & 15;

    for (int kt = 0; kt <= qt; ++kt) {
        __syncthreads();
        for (int c = t; c < 512; c += 128) {
            int row = c >> 3, cc = c & 7;
            async_copy_b128(lds_off32(Ks + row * LDT + cc * 8),
                            kh + (size_t)(kt * 64 + row) * 64 + cc * 8);
            async_copy_b128(lds_off32(Vs + row * LDT + cc * 8),
                            vh + (size_t)(kt * 64 + row) * 64 + cc * 8);
        }
        wait_async0();
        __syncthreads();

        // S = Q K^T : wave handles rows [wv*16, +16) x 64 kv cols.
        v16bf a0 = load_frag_row(Qs + (wv * 16) * LDT + 0,  LDT);
        v16bf a1 = load_frag_row(Qs + (wv * 16) * LDT + 32, LDT);
        v8f sacc[4];
#pragma unroll
        for (int nt = 0; nt < 4; ++nt) {
            sacc[nt] = vzero8();
            v16bf b0 = load_frag_row(Ks + (nt * 16) * LDT + 0,  LDT);
            v16bf b1 = load_frag_row(Ks + (nt * 16) * LDT + 32, LDT);
            sacc[nt] = wmma_bf16(a0, b0, sacc[nt]);
            sacc[nt] = wmma_bf16(a1, b1, sacc[nt]);
        }

        // Online softmax.
        const bool diag = (kt == qt);
        float pv[4][8];
#pragma unroll
        for (int r = 0; r < 8; ++r) {
            int gm = qrow0 + wv * 16 + mbase + r;
            float rmax = -3.0e38f;
#pragma unroll
            for (int nt = 0; nt < 4; ++nt) {
                float v = sacc[nt][r] * scale;
                if (diag && (kt * 64 + nt * 16 + nloc) > gm) v = -3.0e38f;
                pv[nt][r] = v;
                rmax = fmaxf(rmax, v);
            }
            rmax = redmax16(rmax);
            float mnew = fmaxf(mrow[r], rmax);
            float sf   = __expf(mrow[r] - mnew);
            mrow[r] = mnew;
            float rs = 0.f;
#pragma unroll
            for (int nt = 0; nt < 4; ++nt) {
                float p = __expf(pv[nt][r] - mnew);
                pv[nt][r] = p;
                rs += p;
            }
            rs = redsum16(rs);
            lrow[r] = lrow[r] * sf + rs;
#pragma unroll
            for (int dt = 0; dt < 4; ++dt) o[dt][r] *= sf;
        }

        // Re-shape P (C layout) -> A layout via per-wave LDS region.
#pragma unroll
        for (int nt = 0; nt < 4; ++nt)
#pragma unroll
            for (int r = 0; r < 8; ++r)
                Ps[(wv * 16 + mbase + r) * 64 + nt * 16 + nloc] = f2bf(pv[nt][r]);

        // O += P V  (P: 16x64, V: 64x64 K-major in LDS).
#pragma unroll
        for (int dt = 0; dt < 4; ++dt) {
#pragma unroll
            for (int kk = 0; kk < 64; kk += 32) {
                v16bf ap = load_frag_row(Ps + (wv * 16) * 64 + kk, 64);
                v16bf bv = load_frag_kmajor(Vs + kk * LDT + dt * 16, LDT);
                o[dt] = wmma_bf16(ap, bv, o[dt]);
            }
        }
    }

    // Epilogue: out[b, s=m, h*64+d] = o / l   (bf16)
#pragma unroll
    for (int r = 0; r < 8; ++r) {
        int m = qrow0 + wv * 16 + mbase + r;
        float inv = 1.0f / lrow[r];
#pragma unroll
        for (int dt = 0; dt < 4; ++dt) {
            int d = dt * 16 + nloc;
            Ob[((size_t)b * S + m) * 1024 + h * 64 + d] = f2bf(o[dt][r] * inv);
        }
    }
}

// ---------------------------------------------------------------------------
// Host-side orchestration
// ---------------------------------------------------------------------------
extern "C" void kernel_launch(void* const* d_in, const int* in_sizes, int n_in,
                              void* d_out, int out_size, void* d_ws, size_t ws_size,
                              hipStream_t stream) {
    constexpr int B = 2, S = 2048, D = 1024, H = 16, HID = 4096;
    constexpr int M = B * S;  // 4096 rows

    const float* x      = (const float*)d_in[0];
    const float* w_qkv  = (const float*)d_in[1];
    const float* b_qkv  = (const float*)d_in[2];
    const float* w_proj = (const float*)d_in[3];
    const float* b_proj = (const float*)d_in[4];
    const float* g1     = (const float*)d_in[5];
    const float* beta1  = (const float*)d_in[6];
    const float* g2     = (const float*)d_in[7];
    const float* beta2  = (const float*)d_in[8];
    const float* w_fc1  = (const float*)d_in[9];
    const float* b_fc1  = (const float*)d_in[10];
    const float* w_fc2  = (const float*)d_in[11];
    const float* b_fc2  = (const float*)d_in[12];
    float* out          = (float*)d_out;

    char* ws = (char*)d_ws;
    size_t off = 0;
    auto alloc = [&](size_t bytes) -> void* {
        void* p = ws + off;
        off += (bytes + 255) & ~(size_t)255;
        return p;
    };
    u16*   ln1   = (u16*)alloc((size_t)M * D * 2);
    u16*   qb    = (u16*)alloc((size_t)M * D * 2);
    u16*   kb    = (u16*)alloc((size_t)M * D * 2);
    u16*   vb    = (u16*)alloc((size_t)M * D * 2);
    u16*   attnb = (u16*)alloc((size_t)M * D * 2);
    float* x1    = (float*)alloc((size_t)M * D * 4);
    u16*   ln2   = (u16*)alloc((size_t)M * D * 2);
    u16*   fc1b  = (u16*)alloc((size_t)M * HID * 2);
    (void)ws_size; (void)in_sizes; (void)n_in; (void)out_size;

    // 1) LN1
    ln_bf16_kernel<<<M, 256, 0, stream>>>(x, g1, beta1, ln1, D);

    // 2) QKV GEMM (4096 x 3072 x 1024) with head-major scatter
    gemm_bf16_kernel<EPI_QKV><<<dim3(3 * D / 128, M / 128), 256, 0, stream>>>(
        ln1, w_qkv, b_qkv, nullptr, nullptr, nullptr, qb, kb, vb, M, 3 * D, D, S);

    // 3) Causal flash attention
    attn_kernel<<<dim3(S / 64, H, B), 128, 0, stream>>>(qb, kb, vb, attnb, S);

    // 4) Output projection + residual -> x1 (fp32)
    gemm_bf16_kernel<EPI_RES><<<dim3(D / 128, M / 128), 256, 0, stream>>>(
        attnb, w_proj, b_proj, x, nullptr, x1, nullptr, nullptr, nullptr, M, D, D, S);

    // 5) LN2
    ln_bf16_kernel<<<M, 256, 0, stream>>>(x1, g2, beta2, ln2, D);

    // 6) FC1 + exact GELU (4096 x 4096 x 1024)
    gemm_bf16_kernel<EPI_GELU><<<dim3(HID / 128, M / 128), 256, 0, stream>>>(
        ln2, w_fc1, b_fc1, nullptr, fc1b, nullptr, nullptr, nullptr, nullptr, M, HID, D, S);

    // 7) FC2 + residual -> d_out (4096 x 1024 x 4096)
    gemm_bf16_kernel<EPI_RES><<<dim3(D / 128, M / 128), 256, 0, stream>>>(
        fc1b, w_fc2, b_fc2, x1, nullptr, out, nullptr, nullptr, nullptr, M, D, HID, S);
}
